// SSM_76544907149937
// MI455X (gfx1250) — matrix operand order repaired and checked by reading, not compiled, not run on hardware
//
#include <hip/hip_runtime.h>

// ============================================================================
// RSSM VAE forward for MI455X (gfx1250).
// All conv/deconv/FC as implicit-im2col bf16 WMMA GEMMs
// (v_wmma_f32_16x16x32_bf16, fp32 accumulate). Block tile 128x64x32, 8 waves
// (wave32) as 4x2, 32x32 per wave = 4 WMMA per K-step. A tile row-major in
// LDS, B (weights) pre-transposed [N,K] in global with padded rows so the
// B panel is staged with GLOBAL_LOAD_ASYNC_TO_LDS_B128 (ASYNCcnt, inline asm;
// capability probed via __has_builtin), else an aligned b128 copy. Fragments
// are two ds_load_b128 each. Compile-time layer shapes (no runtime div).
// ============================================================================

typedef __attribute__((ext_vector_type(16))) __bf16 v16bf;
typedef __attribute__((ext_vector_type(8)))  float  v8f;

#define LOG2PIF 1.8378770664093453f

#define BM 128
#define BN 64
#define BKK 32

#if defined(__has_builtin)
# if __has_builtin(__builtin_amdgcn_global_load_async_to_lds_b128)
#  define ASYNC_LDS 1
# endif
#endif
#ifndef ASYNC_LDS
# define ASYNC_LDS 0
#endif

// ---------------------------------------------------------------------------
// A-operand loaders. ld1: scalar gather. ld8: 8 consecutive k (16B run);
// caller guarantees row < M and k+8 <= K.
// ---------------------------------------------------------------------------
struct DenseA {
  const __bf16* p; int lda;
  __device__ __bf16 ld1(int row, int k) const {
    return p[(long long)row * lda + k];
  }
  __device__ void ld8(int row, int k, __bf16* dst) const {
    if ((lda & 7) == 0) {
      *(uint4*)dst = *(const uint4*)(p + (long long)row * lda + k);
    } else {
      #pragma unroll
      for (int j = 0; j < 8; ++j) dst[j] = p[(long long)row * lda + k + j];
    }
  }
};

// conv k=4, stride=2, pad=1, NHWC input; k index = (kh*4+kw)*Cin + ci
template <int Cin, int IH, int IW, int OH, int OW>
struct ConvA {
  const __bf16* p;
  __device__ __bf16 ld1(int row, int k) const {
    int ow = row % OW; int t = row / OW; int oh = t % OH; int n = t / OH;
    int ci = k % Cin;  int s = k / Cin;  int kw = s & 3;  int kh = s >> 2;
    int ih = oh * 2 - 1 + kh;
    int iw = ow * 2 - 1 + kw;
    if ((unsigned)ih >= (unsigned)IH || (unsigned)iw >= (unsigned)IW)
      return (__bf16)0.0f;
    return p[(((long long)n * IH + ih) * IW + iw) * Cin + ci];
  }
  __device__ void ld8(int row, int k, __bf16* dst) const {
    if (Cin % 8 == 0) {   // whole run shares (kh,kw): one bounds check, b128
      int ow = row % OW; int t = row / OW; int oh = t % OH; int n = t / OH;
      int ci = k % Cin;  int s = k / Cin; int kw = s & 3; int kh = s >> 2;
      int ih = oh * 2 - 1 + kh;
      int iw = ow * 2 - 1 + kw;
      if ((unsigned)ih >= (unsigned)IH || (unsigned)iw >= (unsigned)IW) {
        *(uint4*)dst = uint4{0, 0, 0, 0};
      } else {
        *(uint4*)dst =
            *(const uint4*)(p + (((long long)n * IH + ih) * IW + iw) * Cin + ci);
      }
    } else {
      #pragma unroll
      for (int j = 0; j < 8; ++j) dst[j] = ld1(row, k + j);
    }
  }
};

// transposed conv k=4,s=2,p=1 (lhs-dilated 2, pad 2), NHWC input
template <int Cin, int IH, int IW, int OH, int OW>
struct DeconvA {
  const __bf16* p;
  __device__ __bf16 ld1(int row, int k) const {
    int ow = row % OW; int t = row / OW; int oh = t % OH; int n = t / OH;
    int ci = k % Cin;  int s = k / Cin;  int kw = s & 3;  int kh = s >> 2;
    int yh = oh - 2 + kh;
    int yw = ow - 2 + kw;
    if (yh < 0 || yw < 0 || (yh & 1) || (yw & 1)) return (__bf16)0.0f;
    int ih = yh >> 1, iw = yw >> 1;
    if (ih >= IH || iw >= IW) return (__bf16)0.0f;
    return p[(((long long)n * IH + ih) * IW + iw) * Cin + ci];
  }
  __device__ void ld8(int row, int k, __bf16* dst) const {
    if (Cin % 8 == 0) {
      int ow = row % OW; int t = row / OW; int oh = t % OH; int n = t / OH;
      int ci = k % Cin;  int s = k / Cin; int kw = s & 3; int kh = s >> 2;
      int yh = oh - 2 + kh;
      int yw = ow - 2 + kw;
      bool bad = (yh < 0) | (yw < 0) | (yh & 1) | (yw & 1);
      int ih = yh >> 1, iw = yw >> 1;
      bad |= (ih >= IH) | (iw >= IW);
      if (bad) {
        *(uint4*)dst = uint4{0, 0, 0, 0};
      } else {
        *(uint4*)dst =
            *(const uint4*)(p + (((long long)n * IH + ih) * IW + iw) * Cin + ci);
      }
    } else {
      #pragma unroll
      for (int j = 0; j < 8; ++j) dst[j] = ld1(row, k + j);
    }
  }
};

// two 16B LDS loads -> one 16-element bf16 fragment (VGPR order preserved)
__device__ inline v16bf frag2(const __bf16* base, int elemOff2) {
  union { uint4 u[2]; v16bf v; } t;
  t.u[0] = *(const uint4*)(base);
  t.u[1] = *(const uint4*)(base + elemOff2);
  return t.v;
}

// ---------------------------------------------------------------------------
// Generic WMMA GEMM:  C[M,N] = act(A[M,K] * Bt^T + bias)
//   Bt is [Npad, ldbt] bf16 (pre-transposed weights, zero-padded rows/cols).
// ---------------------------------------------------------------------------
template <typename AL>
__global__ __launch_bounds__(256)
void gemm_wmma(AL al, const __bf16* __restrict__ Bt, int ldbt,
               const float* __restrict__ bias,
               __bf16* __restrict__ outb, float* __restrict__ outf,
               int M, int N, int K, int relu, int tilesN)
{
  __shared__ __align__(16) __bf16 As[BM * BKK];   // [m][k]
  __shared__ __align__(16) __bf16 Bs[BN * BKK];   // [n][k]

  int bid = (int)blockIdx.x;
  int tn = bid % tilesN, tm = bid / tilesN;
  int rowBase = tm * BM, colBase = tn * BN;
  int tid  = threadIdx.x;
  int lane = tid & 31, wave = tid >> 5;
  int wm = (wave & 3) * 32;   // wave M offset within block tile
  int wn = (wave >> 2) * 32;  // wave N offset within block tile

  v8f c00 = {}, c01 = {}, c10 = {}, c11 = {};

  // staging coordinates (fixed per thread)
  int arow0 = tid >> 2;              // A: rows tid/4 and tid/4+64, chunk tid%4
  int akc   = (tid & 3) * 8;
  int bn    = tid & 63;              // B: row n, 16B chunk tid/64
  int bkc   = (tid >> 6) * 8;
  const __bf16* brow = Bt + (long long)(colBase + bn) * ldbt + bkc;
  __bf16*       bdst = Bs + bn * BKK + bkc;
#if ASYNC_LDS
  __attribute__((address_space(3))) __bf16* bdst3 =
      (__attribute__((address_space(3))) __bf16*)bdst;
#endif

  for (int k0 = 0; k0 < K; k0 += BKK) {
    // ---- stage B panel first (async copy overlaps the A im2col gather) ----
#if ASYNC_LDS
    asm volatile("global_load_async_to_lds_b128 %0, %1, off"
                 :: "v"(bdst3), "v"(brow + k0) : "memory");
#else
    *(uint4*)bdst = *(const uint4*)(brow + k0);
#endif
    // ---- stage A tile (implicit im2col, 16B runs) ----
    #pragma unroll
    for (int half = 0; half < 2; ++half) {
      int r   = arow0 + half * 64;
      int row = rowBase + r;
      int k   = k0 + akc;
      __bf16* dst = As + r * BKK + akc;
      if (row < M && k + 8 <= K) {
        al.ld8(row, k, dst);
      } else {
        #pragma unroll
        for (int j = 0; j < 8; ++j)
          dst[j] = (row < M && k + j < K) ? al.ld1(row, k + j) : (__bf16)0.0f;
      }
    }
#if ASYNC_LDS
    asm volatile("s_wait_asynccnt 0" ::: "memory");
#endif
    __syncthreads();

    // ---- fragments: two ds_load_b128 each ----
    int kbA = (lane >> 4) * 8;   // A halves: K+0..7 / K+8..15 (+16 block)
    int mm0 = wm + (lane & 15);
    v16bf a0 = frag2(As + mm0 * BKK + kbA, 16);
    v16bf a1 = frag2(As + (mm0 + 16) * BKK + kbA, 16);
    int kbB = (lane >> 4) * 8;   // B halves: K+0..7 / K+8..15 (+16 block)
    int nn0 = wn + (lane & 15);
    v16bf b0 = frag2(Bs + nn0 * BKK + kbB, 16);
    v16bf b1 = frag2(Bs + (nn0 + 16) * BKK + kbB, 16);

    c00 = __builtin_amdgcn_wmma_f32_16x16x32_bf16(false, a0, false, b0,
                                                  (short)0, c00, false, false);
    c01 = __builtin_amdgcn_wmma_f32_16x16x32_bf16(false, a0, false, b1,
                                                  (short)0, c01, false, false);
    c10 = __builtin_amdgcn_wmma_f32_16x16x32_bf16(false, a1, false, b0,
                                                  (short)0, c10, false, false);
    c11 = __builtin_amdgcn_wmma_f32_16x16x32_bf16(false, a1, false, b1,
                                                  (short)0, c11, false, false);
    __syncthreads();
  }

  // ---- epilogue: bias + optional ReLU, bf16 and/or fp32 stores ----
  int colA = colBase + wn + (lane & 15);
  int colB = colA + 16;
  int m0   = rowBase + wm + 8 * (lane >> 4);
  float biasA = (bias && colA < N) ? bias[colA] : 0.0f;
  float biasB = (bias && colB < N) ? bias[colB] : 0.0f;

  #pragma unroll
  for (int r = 0; r < 8; ++r) {
    int rowLo = m0 + r, rowHi = m0 + 16 + r;
    if (rowLo < M) {
      if (colA < N) {
        float v = c00[r] + biasA;
        if (relu && v < 0.0f) v = 0.0f;
        if (outb) outb[(long long)rowLo * N + colA] = (__bf16)v;
        if (outf) outf[(long long)rowLo * N + colA] = v;
      }
      if (colB < N) {
        float v = c01[r] + biasB;
        if (relu && v < 0.0f) v = 0.0f;
        if (outb) outb[(long long)rowLo * N + colB] = (__bf16)v;
        if (outf) outf[(long long)rowLo * N + colB] = v;
      }
    }
    if (rowHi < M) {
      if (colA < N) {
        float v = c10[r] + biasA;
        if (relu && v < 0.0f) v = 0.0f;
        if (outb) outb[(long long)rowHi * N + colA] = (__bf16)v;
        if (outf) outf[(long long)rowHi * N + colA] = v;
      }
      if (colB < N) {
        float v = c11[r] + biasB;
        if (relu && v < 0.0f) v = 0.0f;
        if (outb) outb[(long long)rowHi * N + colB] = (__bf16)v;
        if (outf) outf[(long long)rowHi * N + colB] = v;
      }
    }
  }
}

// ---------------------------------------------------------------------------
// weight / data repack kernels (all emit [Npad, ldbt] bf16, zero-padded)
// ---------------------------------------------------------------------------

// OIHW fp32 -> Bt[co][ (kh*4+kw)*Cin+ci ]
__global__ void conv_w_t(const float* __restrict__ w, __bf16* __restrict__ wt,
                         int Cout, int Cin, int Npad, int ldbt) {
  int total = Npad * ldbt;
  int i = blockIdx.x * 256 + threadIdx.x;
  if (i >= total) return;
  int co = i / ldbt; int k = i % ldbt;
  __bf16 v = (__bf16)0.0f;
  if (co < Cout && k < Cin * 16) {
    int ci = k % Cin; int s = k / Cin; int kw = s & 3; int kh = s >> 2;
    v = (__bf16)w[((co * Cin + ci) * 4 + kh) * 4 + kw];
  }
  wt[i] = v;
}

// dense [K,N] fp32 -> Bt[n][k]
__global__ void wT_dense(const float* __restrict__ w, __bf16* __restrict__ wt,
                         int K, int N, int Npad, int ldbt) {
  int total = Npad * ldbt;
  int i = blockIdx.x * 256 + threadIdx.x;
  if (i >= total) return;
  int n = i / ldbt, k = i % ldbt;
  wt[i] = (n < N && k < K) ? (__bf16)w[(long long)k * N + n] : (__bf16)0.0f;
}

// efw [4096,1024], rows permuted NCHW(c*16+s)->NHWC(s*256+c) -> Bt[n][k]
__global__ void efw_t_k(const float* __restrict__ w, __bf16* __restrict__ wt) {
  int i = blockIdx.x * 256 + threadIdx.x;
  if (i >= 1024 * 4096) return;
  int n = i >> 12; int k = i & 4095;
  int s = k >> 8, c = k & 255;
  wt[i] = (__bf16)w[((c * 16 + s) << 10) + n];
}

// dfw [64,4096], cols permuted -> Bt[n=s*256+c][k]
__global__ void dfw_t_k(const float* __restrict__ w, __bf16* __restrict__ wt) {
  int i = blockIdx.x * 256 + threadIdx.x;
  if (i >= 4096 * 64) return;
  int n = i >> 6; int k = i & 63;
  int s = n >> 8, c = n & 255;
  wt[i] = (__bf16)w[(k << 12) + c * 16 + s];
}

__global__ void dfb_perm_k(const float* __restrict__ b, float* __restrict__ o) {
  int i = blockIdx.x * 256 + threadIdx.x;
  if (i >= 4096) return;
  int s = i >> 8, c = i & 255;
  o[i] = b[c * 16 + s];
}

// (B,T,3,64,64) fp32 + x_0 -> frames f=t*B+b (+ 64 x_0 frames) NHWC bf16
__global__ void x_nhwc_k(const float* __restrict__ x, const float* __restrict__ x0,
                         __bf16* __restrict__ out) {
  long long i = (long long)blockIdx.x * 256 + threadIdx.x;
  if (i >= 1984LL * 12288) return;
  int f = (int)(i / 12288); int r = (int)(i % 12288);
  int c = r % 3; int hw = r / 3;
  float v;
  if (f < 1920) {
    int t = f >> 6, b = f & 63;
    v = x[(((long long)(b * 30 + t) * 3 + c) << 12) + hw];
  } else {
    int b = f - 1920;
    v = x0[(((long long)b * 3 + c) << 12) + hw];
  }
  out[i] = (__bf16)v;
}

__global__ void a_bf_k(const float* __restrict__ a, __bf16* __restrict__ o) {
  int i = blockIdx.x * 256 + threadIdx.x;
  if (i >= 7680) return;
  int j = i & 3; int fb = i >> 2; int t = fb >> 6; int b = fb & 63;
  o[i] = (__bf16)a[((b * 30 + t) << 2) + j];
}

// ---------------------------------------------------------------------------
// scan-step kernels
// ---------------------------------------------------------------------------
__global__ void build_x_k(const float* __restrict__ s, const __bf16* __restrict__ h,
                          const __bf16* __restrict__ a, __bf16* __restrict__ X,
                          int Sc, int Hc, int Ac) {
  int K = Sc + Hc + Ac;
  int i = blockIdx.x * 256 + threadIdx.x;
  if (i >= 64 * K) return;
  int b = i / K, k = i % K;
  __bf16 v;
  if (k < Sc)            v = s ? (__bf16)s[b * Sc + k] : (__bf16)0.0f;
  else if (k < Sc + Hc)  v = h[b * Hc + (k - Sc)];
  else                   v = a ? a[b * Ac + (k - Sc - Hc)] : (__bf16)0.0f;
  X[i] = v;
}

__global__ void softplus_k(float* __restrict__ v, int n) {
  int i = blockIdx.x * 256 + threadIdx.x;
  if (i >= n) return;
  float x = v[i];
  v[i] = (x > 20.0f ? x : log1pf(expf(x))) + 0.01f;
}

__global__ void copyf_k(const float* __restrict__ s, float* __restrict__ d, int n) {
  int i = blockIdx.x * 256 + threadIdx.x;
  if (i < n) d[i] = s[i];
}

__global__ void sample_k(const float* __restrict__ loc, const float* __restrict__ sc,
                         const float* __restrict__ eps, float* __restrict__ sprev,
                         __bf16* __restrict__ Sout) {
  int i = blockIdx.x * 256 + threadIdx.x;
  if (i >= 4096) return;
  float v = loc[i] + sc[i] * eps[i];
  if (sprev) sprev[i] = v;
  Sout[i] = (__bf16)v;
}

__global__ void kl01_k(const float* __restrict__ ql, const float* __restrict__ qs,
                       float* __restrict__ acc) {
  float s = 0.0f;
  for (int i = threadIdx.x; i < 4096; i += 256) {
    float l = ql[i], c = qs[i];
    s += -logf(c) + (c * c + l * l) * 0.5f - 0.5f;
  }
  __shared__ float red[256];
  red[threadIdx.x] = s; __syncthreads();
  for (int o = 128; o; o >>= 1) {
    if (threadIdx.x < o) red[threadIdx.x] += red[threadIdx.x + o];
    __syncthreads();
  }
  if (!threadIdx.x) atomicAdd(acc, red[0] * (1.0f / 64.0f));
}

__global__ void klqp_k(const float* __restrict__ ql, const float* __restrict__ qs,
                       const float* __restrict__ pl, const float* __restrict__ ps,
                       float* __restrict__ acc) {
  float s = 0.0f;
  for (int i = threadIdx.x; i < 4096; i += 256) {
    float a = ql[i], b = qs[i], c = pl[i], d = ps[i];
    float dd = a - c;
    s += logf(d / b) + (b * b + dd * dd) / (2.0f * d * d) - 0.5f;
  }
  __shared__ float red[256];
  red[threadIdx.x] = s; __syncthreads();
  for (int o = 128; o; o >>= 1) {
    if (threadIdx.x < o) red[threadIdx.x] += red[threadIdx.x + o];
    __syncthreads();
  }
  if (!threadIdx.x) atomicAdd(acc, red[0] * (1.0f / 64.0f));
}

// recon NLL: sum over all 1920 frames of 0.5*(x-loc)^2 + 0.5*log(2pi), / B
__global__ void recon_k(const __bf16* __restrict__ loc, const float* __restrict__ x,
                        float* __restrict__ acc) {
  const long long total = 1920LL * 3 * 64 * 64;
  float s = 0.0f;
  for (long long i = (long long)blockIdx.x * blockDim.x + threadIdx.x; i < total;
       i += (long long)gridDim.x * blockDim.x) {
    int f  = (int)(i / 12288);
    int r  = (int)(i % 12288);
    int c  = r / 4096;
    int hw = r % 4096;
    int t = f >> 6, b = f & 63;
    float xv = x[(((long long)(b * 30 + t) * 3 + c) << 12) + hw];
    float lv = (float)loc[((long long)f * 4096 + hw) * 3 + c];
    float d = xv - lv;
    s += 0.5f * d * d + 0.5f * LOG2PIF;
  }
  __shared__ float red[256];
  red[threadIdx.x] = s; __syncthreads();
  for (int o = 128; o; o >>= 1) {
    if (threadIdx.x < o) red[threadIdx.x] += red[threadIdx.x + o];
    __syncthreads();
  }
  if (!threadIdx.x) atomicAdd(acc, red[0] * (1.0f / 64.0f));
}

__global__ void init_acc_k(float* acc) {
  if (threadIdx.x < 8) acc[threadIdx.x] = 0.0f;
}
__global__ void finalize_k(const float* __restrict__ acc, float* __restrict__ out) {
  out[0] = acc[0] + acc[1] + acc[2] + acc[3];
}

// ---------------------------------------------------------------------------
// host orchestration
// ---------------------------------------------------------------------------
extern "C" void kernel_launch(void* const* d_in, const int* in_sizes, int n_in,
                              void* d_out, int out_size, void* d_ws, size_t ws_size,
                              hipStream_t stream) {
  (void)in_sizes; (void)n_in; (void)out_size; (void)ws_size;
  const float* x0   = (const float*)d_in[0];
  const float* x    = (const float*)d_in[1];
  const float* a    = (const float*)d_in[2];
  const float* epsq = (const float*)d_in[3];
  const float* epsp = (const float*)d_in[4];
  const float *e1w=(const float*)d_in[5],  *e1b=(const float*)d_in[6];
  const float *e2w=(const float*)d_in[7],  *e2b=(const float*)d_in[8];
  const float *e3w=(const float*)d_in[9],  *e3b=(const float*)d_in[10];
  const float *e4w=(const float*)d_in[11], *e4b=(const float*)d_in[12];
  const float *efw=(const float*)d_in[13], *efb=(const float*)d_in[14];
  const float *q1w=(const float*)d_in[15], *q1b=(const float*)d_in[16];
  const float *qlw=(const float*)d_in[17], *qlb=(const float*)d_in[18];
  const float *qsw=(const float*)d_in[19], *qsb=(const float*)d_in[20];
  const float *p1w=(const float*)d_in[21], *p1b=(const float*)d_in[22];
  const float *plw=(const float*)d_in[23], *plb=(const float*)d_in[24];
  const float *psw=(const float*)d_in[25], *psb=(const float*)d_in[26];
  const float *dfw=(const float*)d_in[27], *dfb=(const float*)d_in[28];
  const float *d1w=(const float*)d_in[29], *d1b=(const float*)d_in[30];
  const float *d2w=(const float*)d_in[31], *d2b=(const float*)d_in[32];
  const float *d3w=(const float*)d_in[33], *d3b=(const float*)d_in[34];
  const float *d4w=(const float*)d_in[35], *d4b=(const float*)d_in[36];

  const int NF = 1984, ND = 1920;

  // workspace carve-up
  char* ws = (char*)d_ws; size_t off = 0;
  auto abf = [&](size_t n) -> __bf16* {
    __bf16* p = (__bf16*)(ws + off); off += ((2 * n + 255) / 256) * 256; return p; };
  auto af32 = [&](size_t n) -> float* {
    float* p = (float*)(ws + off); off += ((4 * n + 255) / 256) * 256; return p; };

  __bf16 *R1 = abf(65011712);  // big ping (enc conv1 out / dec g1,g3)
  __bf16 *R2 = abf(32505856);  // x NHWC / conv2 / conv4 / dec g2 / loc
  __bf16 *R3 = abf(16252928);  // conv3 / dec f0
  // weights, pre-transposed [Npad, ldbt]
  __bf16 *e1wT=abf(64*64),     *e2wT=abf(64*512);
  __bf16 *e3wT=abf(128*1024),  *e4wT=abf(256*2048);
  __bf16 *efwT=abf(1024*4096);
  __bf16 *q1wB=abf(256*1344),  *qlwB=abf(64*256), *qswB=abf(64*256);
  __bf16 *p1wB=abf(256*96),    *plwB=abf(64*256), *pswB=abf(64*256);
  __bf16 *dfwT=abf(4096*64),   *d1wT=abf(128*4096), *d2wT=abf(64*2048);
  __bf16 *d3wT=abf(64*1024),   *d4wT=abf(64*512);
  __bf16 *hall=abf((size_t)NF*1024), *aB=abf(ND*4);
  __bf16 *Sq=abf(ND*64), *Sp=abf(ND*64);
  __bf16 *Xq=abf(64*1344), *Xp=abf(64*68), *hq=abf(64*256), *hp=abf(64*256);
  float *ql=af32(4096), *qs=af32(4096), *pl=af32(4096), *ps=af32(4096);
  float *sprev=af32(4096), *dfbP=af32(4096), *acc=af32(8);

  auto blocks = [](long long n) { return (unsigned)((n + 255) / 256); };

  auto gemm = [&](auto al, const __bf16* Bt, int ldbt, const float* bias,
                  __bf16* ob, float* of, int M, int N, int K, int relu) {
    int tilesN = (N + BN - 1) / BN;
    int tilesM = (M + BM - 1) / BM;
    gemm_wmma<decltype(al)><<<dim3((unsigned)((long long)tilesM * tilesN)),
                              dim3(256), 0, stream>>>(
        al, Bt, ldbt, bias, ob, of, M, N, K, relu, tilesN);
  };

  // --- init + weight repack -------------------------------------------------
  init_acc_k<<<1, 8, 0, stream>>>(acc);
  conv_w_t<<<blocks(64*64),      256, 0, stream>>>(e1w, e1wT,  32,   3,  64,   64);
  conv_w_t<<<blocks(64*512),     256, 0, stream>>>(e2w, e2wT,  64,  32,  64,  512);
  conv_w_t<<<blocks(128*1024),   256, 0, stream>>>(e3w, e3wT, 128,  64, 128, 1024);
  conv_w_t<<<blocks(256*2048),   256, 0, stream>>>(e4w, e4wT, 256, 128, 256, 2048);
  efw_t_k <<<blocks(1024*4096),  256, 0, stream>>>(efw, efwT);
  wT_dense<<<blocks(256*1344),   256, 0, stream>>>(q1w, q1wB, 1344, 256, 256, 1344);
  wT_dense<<<blocks(64*256),     256, 0, stream>>>(qlw, qlwB,  256,  64,  64,  256);
  wT_dense<<<blocks(64*256),     256, 0, stream>>>(qsw, qswB,  256,  64,  64,  256);
  wT_dense<<<blocks(256*96),     256, 0, stream>>>(p1w, p1wB,   68, 256, 256,   96);
  wT_dense<<<blocks(64*256),     256, 0, stream>>>(plw, plwB,  256,  64,  64,  256);
  wT_dense<<<blocks(64*256),     256, 0, stream>>>(psw, pswB,  256,  64,  64,  256);
  dfw_t_k <<<blocks(4096*64),    256, 0, stream>>>(dfw, dfwT);
  dfb_perm_k<<<blocks(4096),     256, 0, stream>>>(dfb, dfbP);
  conv_w_t<<<blocks(128*4096),   256, 0, stream>>>(d1w, d1wT, 128, 256, 128, 4096);
  conv_w_t<<<blocks(64*2048),    256, 0, stream>>>(d2w, d2wT,  64, 128,  64, 2048);
  conv_w_t<<<blocks(64*1024),    256, 0, stream>>>(d3w, d3wT,  32,  64,  64, 1024);
  conv_w_t<<<blocks(64*512),     256, 0, stream>>>(d4w, d4wT,   3,  32,  64,  512);
  x_nhwc_k<<<blocks(1984LL*12288), 256, 0, stream>>>(x, x0, R2);
  a_bf_k  <<<blocks(7680),       256, 0, stream>>>(a, aB);

  // --- encoder (all 1984 frames batched) -----------------------------------
  gemm(ConvA<  3, 64, 64, 32, 32>{R2}, e1wT,   64, e1b, R1, nullptr, NF*1024,  32,   48, 1);
  gemm(ConvA< 32, 32, 32, 16, 16>{R1}, e2wT,  512, e2b, R2, nullptr, NF*256,   64,  512, 1);
  gemm(ConvA< 64, 16, 16,  8,  8>{R2}, e3wT, 1024, e3b, R3, nullptr, NF*64,   128, 1024, 1);
  gemm(ConvA<128,  8,  8,  4,  4>{R3}, e4wT, 2048, e4b, R2, nullptr, NF*16,   256, 2048, 1);
  gemm(DenseA{R2, 4096},               efwT, 4096, efb, hall, nullptr, NF,   1024, 4096, 0);

  // --- s_0: posterior mean with zero state / zero action --------------------
  build_x_k<<<blocks(64*1344), 256, 0, stream>>>(nullptr, hall + 1920*1024, nullptr,
                                                 Xq, 64, 1024, 4);
  gemm(DenseA{Xq, 1344}, q1wB, 1344, q1b, hq, nullptr, 64, 256, 1344, 1);
  gemm(DenseA{hq, 256},  qlwB,  256, qlb, nullptr, ql, 64,  64,  256, 0);
  copyf_k<<<blocks(4096), 256, 0, stream>>>(ql, sprev, 4096);

  // --- latent scan (sequential, carries s_q) --------------------------------
  for (int t = 0; t < 30; ++t) {
    build_x_k<<<blocks(64*1344), 256, 0, stream>>>(sprev, hall + (size_t)t*64*1024,
                                                   aB + t*64*4, Xq, 64, 1024, 4);
    build_x_k<<<blocks(64*68),   256, 0, stream>>>(sprev, nullptr,
                                                   aB + t*64*4, Xp, 64, 0, 4);
    gemm(DenseA{Xq, 1344}, q1wB, 1344, q1b, hq, nullptr, 64, 256, 1344, 1);
    gemm(DenseA{Xp,   68}, p1wB,   96, p1b, hp, nullptr, 64, 256,   68, 1);
    gemm(DenseA{hq,  256}, qlwB,  256, qlb, nullptr, ql, 64,  64,  256, 0);
    gemm(DenseA{hq,  256}, qswB,  256, qsb, nullptr, qs, 64,  64,  256, 0);
    gemm(DenseA{hp,  256}, plwB,  256, plb, nullptr, pl, 64,  64,  256, 0);
    gemm(DenseA{hp,  256}, pswB,  256, psb, nullptr, ps, 64,  64,  256, 0);
    softplus_k<<<blocks(4096), 256, 0, stream>>>(qs, 4096);
    softplus_k<<<blocks(4096), 256, 0, stream>>>(ps, 4096);
    if (t == 29) klqp_k<<<1, 256, 0, stream>>>(ql, qs, pl, ps, acc + 0);
    kl01_k<<<1, 256, 0, stream>>>(ql, qs, acc + 1);
    sample_k<<<blocks(4096), 256, 0, stream>>>(ql, qs, epsq + (size_t)t*4096,
                                               sprev, Sq + (size_t)t*4096);
    sample_k<<<blocks(4096), 256, 0, stream>>>(pl, ps, epsp + (size_t)t*4096,
                                               nullptr, Sp + (size_t)t*4096);
  }

  // --- decoder + recon NLL, run for s_q and s_p ------------------------------
  for (int br = 0; br < 2; ++br) {
    const __bf16* S = br ? Sp : Sq;
    float* accN = acc + (br ? 3 : 2);
    gemm(DenseA{S, 64},                    dfwT,   64, dfbP, R3, nullptr, ND,      4096,  64, 1);
    gemm(DeconvA<256,  4,  4,  8,  8>{R3}, d1wT, 4096, d1b,  R1, nullptr, ND*64,   128, 4096, 1);
    gemm(DeconvA<128,  8,  8, 16, 16>{R1}, d2wT, 2048, d2b,  R2, nullptr, ND*256,   64, 2048, 1);
    gemm(DeconvA< 64, 16, 16, 32, 32>{R2}, d3wT, 1024, d3b,  R1, nullptr, ND*1024,  32, 1024, 1);
    gemm(DeconvA< 32, 32, 32, 64, 64>{R1}, d4wT,  512, d4b,  R2, nullptr, ND*4096,   3,  512, 0);
    recon_k<<<4096, 256, 0, stream>>>(R2, x, accN);
  }

  finalize_k<<<1, 1, 0, stream>>>(acc, (float*)d_out);
}
